// GatingMechanism_67886253080578
// MI455X (gfx1250) — matrix operand order repaired and checked by reading, not compiled, not run on hardware
//
#include <hip/hip_runtime.h>
#include <hip/hip_bf16.h>

// ---------------------------------------------------------------------------
// GatingMechanism on MI455X (gfx1250, wave32, WMMA).
//
// Pipeline (all bf16 WMMA with fp32 accumulate; fused top-k, scores never hit HBM):
//   1. cvt: fp32 -> bf16 for image/text features and the 3 weight matrices
//   2. img_p = img @ W_img^T        (WMMA bf16, fp32 + bf16 outputs)
//   3. txt_p = txt @ W_txt^T        (WMMA bf16, bf16 output)
//   4. scores+top5+softmax fused    (WMMA bf16, 32 rows/block, A in LDS,
//                                    2 WMMA per B fetch -> 4 GB L2 traffic)
//   5. pooled/enhanced gather       (VALU, tiny)
//   6. out = relu(enh @ W_final^T)  (WMMA bf16, fp32 output)
//
// Workspace: ~155 MB (fits the 192 MB global L2).
// ---------------------------------------------------------------------------

typedef __bf16 bf16_t;
typedef __attribute__((ext_vector_type(8)))  __bf16 v8bf;
typedef __attribute__((ext_vector_type(16))) __bf16 v16bf;
typedef __attribute__((ext_vector_type(8)))  float  v8f;

__device__ __forceinline__ bf16_t f2bf(float f) {
  union { float f; unsigned u; } v; v.f = f;
  unsigned r = v.u + 0x7fffu + ((v.u >> 16) & 1u);   // round-to-nearest-even
  unsigned short h = (unsigned short)(r >> 16);
  return __builtin_bit_cast(bf16_t, h);
}
__device__ __forceinline__ float bf2f(bf16_t b) {
  unsigned short h = __builtin_bit_cast(unsigned short, b);
  union { unsigned u; float f; } v; v.u = ((unsigned)h) << 16;
  return v.f;
}
__device__ __forceinline__ v16bf cat16(v8bf lo, v8bf hi) {
  v16bf r;
#pragma unroll
  for (int i = 0; i < 8; ++i) { r[i] = lo[i]; r[i + 8] = hi[i]; }
  return r;
}

// Sorted (descending) top-5 insert with only constant indices (stays in VGPRs).
__device__ __forceinline__ void insert5(float (&tv)[5], int (&ti)[5], float v, int idx) {
  if (v <= tv[4]) return;
  if (v > tv[0]) {
    tv[4]=tv[3]; ti[4]=ti[3]; tv[3]=tv[2]; ti[3]=ti[2];
    tv[2]=tv[1]; ti[2]=ti[1]; tv[1]=tv[0]; ti[1]=ti[0];
    tv[0]=v; ti[0]=idx;
  } else if (v > tv[1]) {
    tv[4]=tv[3]; ti[4]=ti[3]; tv[3]=tv[2]; ti[3]=ti[2];
    tv[2]=tv[1]; ti[2]=ti[1]; tv[1]=v; ti[1]=idx;
  } else if (v > tv[2]) {
    tv[4]=tv[3]; ti[4]=ti[3]; tv[3]=tv[2]; ti[3]=ti[2];
    tv[2]=v; ti[2]=idx;
  } else if (v > tv[3]) {
    tv[4]=tv[3]; ti[4]=ti[3]; tv[3]=v; ti[3]=idx;
  } else {
    tv[4]=v; ti[4]=idx;
  }
}

// ---------------------------------------------------------------------------
// fp32 -> bf16 conversion, 8 elements/thread, b128 loads.
// ---------------------------------------------------------------------------
__global__ __launch_bounds__(256) void cvt_f32_bf16(const float* __restrict__ in,
                                                    bf16_t* __restrict__ out,
                                                    long long n) {
  long long i = ((long long)blockIdx.x * 256 + threadIdx.x) * 8;
  if (i + 8 <= n) {
    float4 a = *(const float4*)(in + i);
    float4 b = *(const float4*)(in + i + 4);
    v8bf o;
    o[0]=f2bf(a.x); o[1]=f2bf(a.y); o[2]=f2bf(a.z); o[3]=f2bf(a.w);
    o[4]=f2bf(b.x); o[5]=f2bf(b.y); o[6]=f2bf(b.z); o[7]=f2bf(b.w);
    *(v8bf*)(out + i) = o;
  }
}

// ---------------------------------------------------------------------------
// Generic out[M,1024] = X[M,1024] @ W[1024,1024]^T  (torch Linear semantics)
// Block = 256 threads = 8 waves; wave -> 16x64 tile (4 WMMA accumulators).
// Grid = (1024/64, M/128).
// Fragment layouts per CDNA5 ISA 7.12.2:
//   A 16x32 bf16: lane l holds row l%16; K-halves split by lane>=16.
//   B 32x16 bf16: lane l holds col l%16; same K order as A (k0+8*half.., k0+16+8*half..).
// ---------------------------------------------------------------------------
__global__ __launch_bounds__(256) void gemm_xwt_bf16(const bf16_t* __restrict__ X,
                                                     const bf16_t* __restrict__ W,
                                                     float* __restrict__ outF,
                                                     bf16_t* __restrict__ outB,
                                                     int M, int relu) {
  const int lane = threadIdx.x & 31;
  const int wv   = threadIdx.x >> 5;
  const int half = lane >> 4;     // 0: lanes 0-15, 1: lanes 16-31
  const int l16  = lane & 15;
  const int mbase = blockIdx.y * 128 + wv * 16;
  const int nbase = blockIdx.x * 64;

  const size_t arow = (size_t)(mbase + l16) * 1024;
  v8f acc[4] = {{}, {}, {}, {}};

  for (int kt = 0; kt < 32; ++kt) {
    const int k0 = kt * 32;
    v8bf alo = *(const v8bf*)(X + arow + k0 + half * 8);
    v8bf ahi = *(const v8bf*)(X + arow + k0 + 16 + half * 8);
    v16bf A = cat16(alo, ahi);
#pragma unroll
    for (int t = 0; t < 4; ++t) {
      const size_t brow = (size_t)(nbase + t * 16 + l16) * 1024;
      v8bf blo = *(const v8bf*)(W + brow + k0 + half * 8);
      v8bf bhi = *(const v8bf*)(W + brow + k0 + 16 + half * 8);
      acc[t] = __builtin_amdgcn_wmma_f32_16x16x32_bf16(
          false, A, false, cat16(blo, bhi), (short)0, acc[t], false, false);
    }
  }

  // C layout: VGPR r -> row mbase + r + 8*half, col nbase + t*16 + l16
#pragma unroll
  for (int t = 0; t < 4; ++t) {
    const int col = nbase + t * 16 + l16;
#pragma unroll
    for (int r = 0; r < 8; ++r) {
      const size_t m = (size_t)(mbase + r + 8 * half);
      float v = acc[t][r];
      if (relu) v = v > 0.f ? v : 0.f;
      if (outF) outF[m * 1024 + col] = v;
      if (outB) outB[m * 1024 + col] = f2bf(v);
    }
  }
}

// ---------------------------------------------------------------------------
// Fused scores = img_p @ txt_p^T  +  top-5  +  softmax weights.
// One block per 32 image rows (A tile 32x1024 bf16 in dynamic LDS, padded
// stride). Each wave owns 16 columns per N-chunk and computes BOTH 16-row
// tiles from one B fragment fetch (2 WMMA per B load -> 2x arithmetic
// intensity vs txt_p, ~4 GB total L2 traffic). txt_p streamed with prefetch.
// Top-5: one row per lane (all 32 lanes busy), cross-wave merge in LDS.
//
// Dynamic LDS carve (bytes):
//   sA : 32*1032*2          = 66048
//   sC : 8 waves*32*16*4    = 16384   (off 66048)
//   sMv: 8*32*5*4           =  5120   (off 82432)
//   sMi: 8*32*5*4           =  5120   (off 87552)   total 92672
// ---------------------------------------------------------------------------
#define SCORES_LDS_BYTES 92672

__global__ __launch_bounds__(256) void scores_topk_kernel(const bf16_t* __restrict__ imgp,
                                                          const bf16_t* __restrict__ txtp,
                                                          float* __restrict__ topw,
                                                          int* __restrict__ topi,
                                                          int N) {
  extern __shared__ char smem[];
  bf16_t* sA  = (bf16_t*)smem;               // [32][1032]
  float*  sC  = (float*)(smem + 66048);      // [8][32][16]
  float*  sMv = (float*)(smem + 82432);      // [8][32][5]
  int*    sMi = (int*)  (smem + 87552);      // [8][32][5]

  const int tid  = threadIdx.x;
  const int lane = tid & 31;
  const int wv   = tid >> 5;
  const int half = lane >> 4;
  const int l16  = lane & 15;
  const int mbase = blockIdx.x * 32;

  // Stage the 32x1024 bf16 img_p tile (coalesced b128 loads/stores).
  for (int ve = tid; ve < 4096; ve += 256) {
    const int r = ve >> 7;            // 128 vec8 per row
    const int c = (ve & 127) * 8;
    *(v8bf*)&sA[r * 1032 + c] = *(const v8bf*)(imgp + (size_t)(mbase + r) * 1024 + c);
  }
  __syncthreads();

  float tv[5]; int ti[5];
#pragma unroll
  for (int k = 0; k < 5; ++k) { tv[k] = -3.4e38f; ti[k] = 0; }

  for (int nb = wv * 16; nb < N; nb += 128) {
    const size_t brow = (size_t)(nb + l16) * 1024 + half * 8;
    if (nb + 128 < N) __builtin_prefetch(txtp + brow + (size_t)128 * 1024, 0, 0);

    v8f acc0 = {}, acc1 = {};
#pragma unroll 4
    for (int kt = 0; kt < 32; ++kt) {
      const int k0 = kt * 32;
      // one B fragment, two A row-tiles
      v8bf blo = *(const v8bf*)(txtp + brow + k0);
      v8bf bhi = *(const v8bf*)(txtp + brow + k0 + 16);
      v16bf Bf = cat16(blo, bhi);
      v8bf a0lo = *(const v8bf*)&sA[l16 * 1032 + k0 + half * 8];
      v8bf a0hi = *(const v8bf*)&sA[l16 * 1032 + k0 + 16 + half * 8];
      acc0 = __builtin_amdgcn_wmma_f32_16x16x32_bf16(
          false, cat16(a0lo, a0hi), false, Bf, (short)0, acc0, false, false);
      v8bf a1lo = *(const v8bf*)&sA[(16 + l16) * 1032 + k0 + half * 8];
      v8bf a1hi = *(const v8bf*)&sA[(16 + l16) * 1032 + k0 + 16 + half * 8];
      acc1 = __builtin_amdgcn_wmma_f32_16x16x32_bf16(
          false, cat16(a1lo, a1hi), false, Bf, (short)0, acc1, false, false);
    }

    // Spill both 16x16 tiles: wave scratch is [32 rows][16 cols].
#pragma unroll
    for (int r = 0; r < 8; ++r) {
      sC[(wv * 32 + (r + 8 * half)) * 16 + l16]      = acc0[r];
      sC[(wv * 32 + 16 + (r + 8 * half)) * 16 + l16] = acc1[r];
    }
    asm volatile("s_wait_dscnt 0" ::: "memory");   // wave-local LDS RAW fence

    // All 32 lanes scan: lane owns row `lane` of this wave's 32-row scratch.
#pragma unroll
    for (int c = 0; c < 16; ++c)
      insert5(tv, ti, sC[(wv * 32 + lane) * 16 + c], nb + c);
  }

  // Publish per-wave top-5 (one row per lane) and merge across the 8 waves.
#pragma unroll
  for (int k = 0; k < 5; ++k) {
    sMv[(wv * 32 + lane) * 5 + k] = tv[k];
    sMi[(wv * 32 + lane) * 5 + k] = ti[k];
  }
  __syncthreads();

  if (wv == 0) {   // one full wave: lane merges row mbase+lane
    float fv[5]; int fi[5];
#pragma unroll
    for (int k = 0; k < 5; ++k) { fv[k] = -3.4e38f; fi[k] = 0; }
    for (int w2 = 0; w2 < 8; ++w2)
#pragma unroll
      for (int k = 0; k < 5; ++k)
        insert5(fv, fi, sMv[(w2 * 32 + lane) * 5 + k], sMi[(w2 * 32 + lane) * 5 + k]);

    // softmax over the 5 (sorted desc -> fv[0] is the max)
    float e[5], s = 0.f;
#pragma unroll
    for (int k = 0; k < 5; ++k) { e[k] = __expf(fv[k] - fv[0]); s += e[k]; }
    const float inv = 1.f / s;
    const int row = mbase + lane;
#pragma unroll
    for (int k = 0; k < 5; ++k) { topw[row * 5 + k] = e[k] * inv; topi[row * 5 + k] = fi[k]; }
  }
}

// ---------------------------------------------------------------------------
// enhanced[b,:] = img_p[b,:] * sum_k w_k * txt_p[idx_k,:]   (bf16 out)
// ---------------------------------------------------------------------------
__global__ __launch_bounds__(256) void pool_enhance_kernel(const float* __restrict__ imgp,
                                                           const bf16_t* __restrict__ txtp,
                                                           const float* __restrict__ topw,
                                                           const int* __restrict__ topi,
                                                           bf16_t* __restrict__ enh) {
  const int b = blockIdx.x;
  float w[5]; int id[5];
#pragma unroll
  for (int k = 0; k < 5; ++k) { w[k] = topw[b * 5 + k]; id[k] = topi[b * 5 + k]; }
  for (int c = threadIdx.x; c < 1024; c += 256) {
    float p = 0.f;
#pragma unroll
    for (int k = 0; k < 5; ++k) p += w[k] * bf2f(txtp[(size_t)id[k] * 1024 + c]);
    enh[(size_t)b * 1024 + c] = f2bf(imgp[(size_t)b * 1024 + c] * p);
  }
}

// ---------------------------------------------------------------------------
extern "C" void kernel_launch(void* const* d_in, const int* in_sizes, int n_in,
                              void* d_out, int out_size, void* d_ws, size_t ws_size,
                              hipStream_t stream) {
  const float* imgF = (const float*)d_in[0];   // [B,1024]
  const float* txtF = (const float*)d_in[1];   // [N,1024]
  const float* Wimg = (const float*)d_in[2];   // [1024,1024]
  const float* Wtxt = (const float*)d_in[3];   // [1024,1024]
  const float* Wfin = (const float*)d_in[4];   // [1024,1024]
  const int B = in_sizes[0] / 1024;            // 2048
  const int N = in_sizes[1] / 1024;            // 32768

  size_t off = 0;
  auto carve = [&](size_t bytes) -> void* {
    void* p = (char*)d_ws + off;
    off = (off + bytes + 255) & ~(size_t)255;
    return p;
  };
  bf16_t* bimg  = (bf16_t*)carve((size_t)B * 1024 * 2);
  bf16_t* btxt  = (bf16_t*)carve((size_t)N * 1024 * 2);
  bf16_t* bWi   = (bf16_t*)carve((size_t)1024 * 1024 * 2);
  bf16_t* bWt   = (bf16_t*)carve((size_t)1024 * 1024 * 2);
  bf16_t* bWf   = (bf16_t*)carve((size_t)1024 * 1024 * 2);
  float*  imgpF = (float*) carve((size_t)B * 1024 * 4);
  bf16_t* imgpB = (bf16_t*)carve((size_t)B * 1024 * 2);
  bf16_t* txtpB = (bf16_t*)carve((size_t)N * 1024 * 2);
  float*  topw  = (float*) carve((size_t)B * 5 * 4);
  int*    topi  = (int*)   carve((size_t)B * 5 * 4);
  bf16_t* enh   = (bf16_t*)carve((size_t)B * 1024 * 2);
  (void)ws_size; (void)n_in; (void)out_size;

  // 1. fp32 -> bf16
  cvt_f32_bf16<<<(int)(((long long)B * 1024) / 2048), 256, 0, stream>>>(imgF, bimg, (long long)B * 1024);
  cvt_f32_bf16<<<(int)(((long long)N * 1024) / 2048), 256, 0, stream>>>(txtF, btxt, (long long)N * 1024);
  cvt_f32_bf16<<<512, 256, 0, stream>>>(Wimg, bWi, 1024LL * 1024);
  cvt_f32_bf16<<<512, 256, 0, stream>>>(Wtxt, bWt, 1024LL * 1024);
  cvt_f32_bf16<<<512, 256, 0, stream>>>(Wfin, bWf, 1024LL * 1024);

  // 2/3. projections
  gemm_xwt_bf16<<<dim3(16, B / 128), 256, 0, stream>>>(bimg, bWi, imgpF, imgpB, B, 0);
  gemm_xwt_bf16<<<dim3(16, N / 128), 256, 0, stream>>>(btxt, bWt, (float*)nullptr, txtpB, N, 0);

  // 4. fused scores + top-5 + softmax (32 rows/block, dynamic LDS)
  scores_topk_kernel<<<B / 32, 256, SCORES_LDS_BYTES, stream>>>(imgpB, txtpB, topw, topi, N);

  // 5. pooled / enhanced
  pool_enhance_kernel<<<B, 256, 0, stream>>>(imgpF, txtpB, topw, topi, enh);

  // 6. out = relu(enh @ W_final^T)
  gemm_xwt_bf16<<<dim3(16, B / 128), 256, 0, stream>>>(enh, bWf, (float*)d_out, (bf16_t*)nullptr, B, 1);
}
